// HomoMoleculeGNN_GAT_18013092839577
// MI455X (gfx1250) — compile-verified, matrix operand order checked by reading, not run on hardware
//
#include <hip/hip_runtime.h>
#include <hip/hip_bf16.h>

typedef __attribute__((ext_vector_type(16))) _Float16 v16h;
typedef __attribute__((ext_vector_type(8)))  float    v8f;

// ---------------------------------------------------------------------------
// Build h0 = concat(ntype_emb[ntypes], x)   (NN x 80)
// ---------------------------------------------------------------------------
__global__ void embed_nodes_kernel(const float* __restrict__ x,
                                   const int* __restrict__ ntypes,
                                   const float* __restrict__ ntype_emb,
                                   float* __restrict__ h, int n) {
  int i = blockIdx.x * blockDim.x + threadIdx.x;
  if (i >= n * 80) return;
  int node = i / 80, c = i % 80;
  h[i] = (c < 16) ? ntype_emb[ntypes[node] * 16 + c]
                  : x[node * 64 + (c - 16)];
}

// ---------------------------------------------------------------------------
// Y[M x 128] = X[M x K] @ W[K x 128] + bias   via v_wmma_f32_16x16x32_f16
// K is a compile-time constant (80 or 64): padding decisions constant-fold,
// fragment loads are unconditional & vectorizable. Block = 256 threads
// (8 waves); block stages its 16-row X slab into LDS as f16 (padded stride,
// conflict-free ds_load_b128 fragment reads); wave w owns a 16-col W slab.
// EXEC stays all-ones around the WMMAs; store epilogue uses one uniform
// branch (fast unguarded path when the whole row-block is in range).
// ---------------------------------------------------------------------------
template <int K>
__global__ __launch_bounds__(256) void gemm_bias_wmma_kernel(
    const float* __restrict__ X, const float* __restrict__ W,
    const float* __restrict__ bias, float* __restrict__ Y, int M) {
  static_assert(K % 16 == 0, "K must be a multiple of 16");
  constexpr int KP  = (K + 31) & ~31;  // padded to WMMA K-step
  constexpr int LDP = KP + 8;          // LDS row stride (52 dwords -> no bank conflicts)
  __shared__ _Float16 Ash[16 * LDP];

  const int tid  = threadIdx.x;
  const int wave = tid >> 5;
  const int lane = tid & 31;
  const int m0   = blockIdx.x << 4;
  const int n0   = wave << 4;
  const int halfsel = (lane >> 4) << 3;    // 0 or 8 (K sub-offset for A/B frags)
  const int col  = n0 + (lane & 15);

  // Cooperative stage: 16 x K f32 tile of X -> f16 in LDS (zero-padded to KP).
#pragma unroll
  for (int idx = tid; idx < 16 * KP; idx += 256) {
    const int r = idx / KP, k = idx % KP;
    const int row = min(m0 + r, M - 1);
    Ash[r * LDP + k] = (k < K) ? (_Float16)X[row * K + k] : (_Float16)0.0f;
  }
  __syncthreads();

  const _Float16* __restrict__ arow = &Ash[(lane & 15) * LDP];
  v8f acc = {};
#pragma unroll
  for (int k0 = 0; k0 < KP; k0 += 32) {
    v16h a, b;
#pragma unroll
    for (int j = 0; j < 16; ++j) {
      const int kb = k0 + ((j >> 3) << 4);       // compile-time constant
      const int k  = kb + halfsel + (j & 7);
      a[j] = arow[k];                            // consecutive halves -> ds_load_b128
      b[j] = (kb < K) ? (_Float16)W[k * 128 + col] : (_Float16)0.0f;
    }
    acc = __builtin_amdgcn_wmma_f32_16x16x32_f16(
        /*neg_a=*/false, a, /*neg_b=*/false, b,
        /*c_mod=*/(short)0, acc, /*reuse_a=*/false, /*reuse_b=*/false);
  }

  const int rbase = m0 + halfsel;   // D: VGPR v -> row rbase+v, col fixed per lane
  const float bb = bias[col];
  if (m0 + 16 <= M) {
    // fast path: whole row-block in range, no per-element predication
    float* __restrict__ yp = Y + (size_t)rbase * 128 + col;
#pragma unroll
    for (int v = 0; v < 8; ++v) yp[v * 128] = acc[v] + bb;
  } else {
#pragma unroll
    for (int v = 0; v < 8; ++v) {
      const int r = rbase + v;
      if (r < M) Y[r * 128 + col] = acc[v] + bb;
    }
  }
}

// ---------------------------------------------------------------------------
// Init per-layer accumulators: agg=0, amax=-inf, denom=0
// ---------------------------------------------------------------------------
__global__ void init_accum_kernel(float* __restrict__ agg,
                                  float* __restrict__ amax,
                                  float* __restrict__ denom,
                                  int nagg, int nmax) {
  int i = blockIdx.x * blockDim.x + threadIdx.x;
  if (i < nagg) agg[i] = 0.0f;
  if (i < nmax) {
    amax[i] = __uint_as_float(0xFF800000u);  // -inf
    denom[i] = 0.0f;
  }
}

// ---------------------------------------------------------------------------
// Per-edge attention logit:
//   ef = concat(etype_emb[etype], eattr) @ We   (inline 24x128 matvec, LDS-staged)
//   z  = leaky_relu(xl[src] + xr[dst] + ef)
//   alpha[e,h] = sum_c z[h,c]*att[h,c];  atomic float-max into amax[dst,h]
// 2 edges per 256-thread block; thread t = h*64 + c.
// ---------------------------------------------------------------------------
__global__ __launch_bounds__(256) void edge_alpha_kernel(
    const int* __restrict__ eidx, const int* __restrict__ etypes,
    const float* __restrict__ eattr, const float* __restrict__ etype_emb,
    const float* __restrict__ We, const float* __restrict__ att,
    const float* __restrict__ xl, const float* __restrict__ xr,
    float* __restrict__ alpha, float* __restrict__ amax, int E) {
  __shared__ float ea_sh[2][24];
  __shared__ float red[256];
  const int tid = threadIdx.x;
  const int el = tid >> 7;          // which edge of this block
  const int t  = tid & 127;         // h*64 + c
  const int e  = blockIdx.x * 2 + el;
  const int ec = min(e, E - 1);     // clamped for uniform control flow

  if (t < 8)        ea_sh[el][t] = etype_emb[etypes[ec] * 8 + t];
  else if (t < 24)  ea_sh[el][t] = eattr[ec * 16 + (t - 8)];
  __syncthreads();

  const int src = eidx[ec];
  const int dst = eidx[E + ec];
  float ef = 0.0f;
#pragma unroll
  for (int k = 0; k < 24; ++k) ef += ea_sh[el][k] * We[k * 128 + t];

  float z = xl[src * 128 + t] + xr[dst * 128 + t] + ef;
  z = (z > 0.0f) ? z : 0.2f * z;
  red[tid] = z * att[t];
  __syncthreads();
#pragma unroll
  for (int s = 32; s >= 1; s >>= 1) {
    if ((t & 63) < s) red[tid] += red[tid + s];
    __syncthreads();
  }
  if (((t & 63) == 0) && (e < E)) {
    const int h = t >> 6;
    const float a = red[tid];
    alpha[e * 2 + h] = a;
    int* addr = (int*)&amax[dst * 2 + h];
    if (a >= 0.0f) atomicMax(addr, __float_as_int(a));
    else           atomicMin((unsigned int*)addr, __float_as_uint(a));
  }
}

// ---------------------------------------------------------------------------
// ex = exp(alpha - amax[dst]);  denom[dst] += ex  (amax -inf -> 0, ref semantics)
// ---------------------------------------------------------------------------
__global__ void edge_exp_kernel(const int* __restrict__ eidx,
                                float* __restrict__ alpha,
                                const float* __restrict__ amax,
                                float* __restrict__ denom, int E) {
  int i = blockIdx.x * blockDim.x + threadIdx.x;  // over 2*E (e,h) pairs
  if (i >= 2 * E) return;
  int e = i >> 1, h = i & 1;
  int dst = eidx[E + e];
  float m = amax[dst * 2 + h];
  if (__float_as_uint(m) == 0xFF800000u) m = 0.0f;  // non-finite -> 0
  float ex = expf(alpha[i] - m);
  alpha[i] = ex;
  atomicAdd(&denom[dst * 2 + h], ex);
}

// ---------------------------------------------------------------------------
// agg[dst,h,c] += xl[src,h,c] * ex / (denom[dst,h] + 1e-16)
// ---------------------------------------------------------------------------
__global__ __launch_bounds__(256) void edge_scatter_kernel(
    const int* __restrict__ eidx, const float* __restrict__ alpha,
    const float* __restrict__ denom, const float* __restrict__ xl,
    float* __restrict__ agg, int E) {
  const int tid = threadIdx.x;
  const int el = tid >> 7;
  const int t  = tid & 127;
  const int e  = blockIdx.x * 2 + el;
  if (e >= E) return;
  const int src = eidx[e];
  const int dst = eidx[E + e];
  const int h = t >> 6;
  const float w = alpha[e * 2 + h] / (denom[dst * 2 + h] + 1e-16f);
  atomicAdd(&agg[dst * 128 + t], xl[src * 128 + t] * w);
}

// ---------------------------------------------------------------------------
// h_out[n,c] = relu( mean_h(agg[n,h,c]) + bias[c] )
// ---------------------------------------------------------------------------
__global__ void node_out_kernel(const float* __restrict__ agg,
                                const float* __restrict__ bias,
                                float* __restrict__ hout, int n) {
  int i = blockIdx.x * blockDim.x + threadIdx.x;
  if (i >= n * 64) return;
  int node = i >> 6, c = i & 63;
  float v = 0.5f * (agg[node * 128 + c] + agg[node * 128 + 64 + c]) + bias[c];
  hout[i] = fmaxf(v, 0.0f);
}

// ---------------------------------------------------------------------------
extern "C" void kernel_launch(void* const* d_in, const int* in_sizes, int n_in,
                              void* d_out, int out_size, void* d_ws, size_t ws_size,
                              hipStream_t stream) {
  const float* x         = (const float*)d_in[0];
  const int*   eidx      = (const int*)  d_in[1];
  const int*   ntypes    = (const int*)  d_in[2];
  const int*   etypes    = (const int*)  d_in[3];
  const float* eattr     = (const float*)d_in[4];
  const float* ntype_emb = (const float*)d_in[5];
  const float* etype_emb = (const float*)d_in[6];

  const int n = in_sizes[2];   // 50000 nodes
  const int E = in_sizes[3];   // 800000 edges

  // workspace layout (floats), ~113 MB total
  float* ws = (float*)d_ws;
  float* hA    = ws;                    // n*80  (h0; then layer-1 output ping-pong)
  float* hB    = hA + (size_t)n * 80;   // n*64
  float* xl    = hB + (size_t)n * 64;   // n*128
  float* xr    = xl + (size_t)n * 128;  // n*128
  float* agg   = xr + (size_t)n * 128;  // n*128
  float* alpha = agg + (size_t)n * 128; // E*2
  float* amax  = alpha + (size_t)E * 2; // n*2
  float* denom = amax + (size_t)n * 2;  // n*2

  const int nagg = n * 128, nmax = n * 2;

  embed_nodes_kernel<<<(n * 80 + 255) / 256, 256, 0, stream>>>(
      x, ntypes, ntype_emb, hA, n);

  for (int l = 0; l < 3; ++l) {
    const float* Wl   = (const float*)d_in[7 + l * 7 + 0];
    const float* bl   = (const float*)d_in[7 + l * 7 + 1];
    const float* Wr   = (const float*)d_in[7 + l * 7 + 2];
    const float* br   = (const float*)d_in[7 + l * 7 + 3];
    const float* We   = (const float*)d_in[7 + l * 7 + 4];
    const float* att  = (const float*)d_in[7 + l * 7 + 5];
    const float* bias = (const float*)d_in[7 + l * 7 + 6];

    const float* Xc = (l == 0) ? hA : ((l == 1) ? hB : hA);
    const int gemm_blocks = (n + 15) / 16;

    if (l == 0) {
      gemm_bias_wmma_kernel<80><<<gemm_blocks, 256, 0, stream>>>(Xc, Wl, bl, xl, n);
      gemm_bias_wmma_kernel<80><<<gemm_blocks, 256, 0, stream>>>(Xc, Wr, br, xr, n);
    } else {
      gemm_bias_wmma_kernel<64><<<gemm_blocks, 256, 0, stream>>>(Xc, Wl, bl, xl, n);
      gemm_bias_wmma_kernel<64><<<gemm_blocks, 256, 0, stream>>>(Xc, Wr, br, xr, n);
    }

    init_accum_kernel<<<(nagg + 255) / 256, 256, 0, stream>>>(
        agg, amax, denom, nagg, nmax);

    edge_alpha_kernel<<<(E + 1) / 2, 256, 0, stream>>>(
        eidx, etypes, eattr, etype_emb, We, att, xl, xr, alpha, amax, E);

    edge_exp_kernel<<<(2 * E + 255) / 256, 256, 0, stream>>>(
        eidx, alpha, amax, denom, E);

    edge_scatter_kernel<<<(E + 1) / 2, 256, 0, stream>>>(
        eidx, alpha, denom, xl, agg, E);

    float* Yout = (l == 2) ? (float*)d_out : ((l == 0) ? hB : hA);
    node_out_kernel<<<(n * 64 + 255) / 256, 256, 0, stream>>>(
        agg, bias, Yout, n);
  }
}
